// RWKVBlock_9792525435423
// MI455X (gfx1250) — matrix operand order repaired and checked by reading, not compiled, not run on hardware
//
#include <hip/hip_runtime.h>
#include <hip/hip_bf16.h>
#include <math.h>

// ---------------------------------------------------------------------------
// Vision-RWKV block for MI455X (gfx1250, wave32, WMMA).
// All heavy GEMMs run on v_wmma_f32_16x16x32_bf16 (fp32 accumulate).
// GEMM uses 64x64 register tiles per wave: 16 WMMA per K-step vs 16 b128
// loads -> 1 load per WMMA (4x operand reuse), easing L0/L1 pressure.
// ---------------------------------------------------------------------------

typedef __attribute__((ext_vector_type(16))) __bf16 v16bf;
typedef __attribute__((ext_vector_type(8)))  __bf16 v8bf;
typedef __attribute__((ext_vector_type(8)))  float  v8f;

// Problem constants (match setup_inputs exactly).
constexpr int   Bsz = 16;
constexpr int   Hh  = 32;
constexpr int   Ww  = 32;
constexpr int   Cc  = 1024;
constexpr int   HID = 4096;
constexpr int   Nn  = Hh * Ww;              // 1024 tokens
constexpr long long BNCll = (long long)Bsz * Nn * Cc;   // 16,777,216
constexpr long long BNHll = (long long)Bsz * Nn * HID;  // 67,108,864

// ---------------------------------------------------------------------------
// fp32 -> bf16 conversion (weights + misc)
// ---------------------------------------------------------------------------
__global__ void cvt_f32_bf16(const float* __restrict__ in,
                             __bf16* __restrict__ out, long long n) {
  long long i = (long long)blockIdx.x * blockDim.x + threadIdx.x;
  if (i < n) out[i] = (__bf16)in[i];
}

// ---------------------------------------------------------------------------
// LayerNorm over last dim (C=1024). One 256-thread block per row.
// ---------------------------------------------------------------------------
__global__ __launch_bounds__(256) void ln_kernel(const float* __restrict__ x,
                                                 const float* __restrict__ g,
                                                 const float* __restrict__ b,
                                                 float* __restrict__ out) {
  const long long row = blockIdx.x;
  const float* xr = x + row * Cc;
  float s = 0.f, s2 = 0.f;
  for (int c = threadIdx.x; c < Cc; c += 256) {
    float v = xr[c];
    s += v; s2 += v * v;
  }
  // wave32 shuffle reduce
  for (int off = 16; off > 0; off >>= 1) {
    s  += __shfl_xor(s,  off, 32);
    s2 += __shfl_xor(s2, off, 32);
  }
  __shared__ float sh[16];
  const int wave = threadIdx.x >> 5, lane = threadIdx.x & 31;
  if (lane == 0) { sh[wave] = s; sh[8 + wave] = s2; }
  __syncthreads();
  float ts = 0.f, ts2 = 0.f;
  for (int i = 0; i < 8; ++i) { ts += sh[i]; ts2 += sh[8 + i]; }
  const float mean = ts * (1.f / Cc);
  const float var  = ts2 * (1.f / Cc) - mean * mean;
  const float rstd = rsqrtf(var + 1e-5f);
  float* orow = out + row * Cc;
  for (int c = threadIdx.x; c < Cc; c += 256)
    orow[c] = (xr[c] - mean) * rstd * g[c] + b[c];
}

// ---------------------------------------------------------------------------
// Multihead 4-direction q-shift + token-mix blend; emits bf16 GEMM operands.
// head_dim=64, gamma group g=16 -> 4 groups cover all channels.
// o2 may be null (channel-mix path only needs two blends).
// ---------------------------------------------------------------------------
__global__ __launch_bounds__(256) void shiftmix_kernel(
    const float* __restrict__ xs,
    const float* __restrict__ mu0, const float* __restrict__ mu1,
    const float* __restrict__ mu2,
    __bf16* __restrict__ o0, __bf16* __restrict__ o1,
    __bf16* __restrict__ o2) {
  long long i = (long long)blockIdx.x * blockDim.x + threadIdx.x;
  if (i >= BNCll) return;
  const int c = (int)(i % Cc);
  const long long bn = i / Cc;
  const int n = (int)(bn % Nn);
  const long long bb = bn / Nn;
  const int y = n / Ww, xw = n % Ww;
  const int ch  = c & 63;     // within head (hd=64)
  const int grp = ch >> 4;    // 16-channel gamma groups
  int sy = y, sx = xw;
  if      (grp == 0) sx = xw - 1;   // from left
  else if (grp == 1) sx = xw + 1;   // from right
  else if (grp == 2) sy = y - 1;    // from above
  else               sy = y + 1;    // from below
  float xx = 0.f;
  if (sx >= 0 && sx < Ww && sy >= 0 && sy < Hh)
    xx = xs[(bb * Nn + (long long)sy * Ww + sx) * Cc + c];
  const float xv = xs[i];
  const float m0 = mu0[c];
  o0[i] = (__bf16)(xv * m0 + xx * (1.f - m0));
  const float m1 = mu1[c];
  o1[i] = (__bf16)(xv * m1 + xx * (1.f - m1));
  if (o2) {
    const float m2 = mu2[c];
    o2[i] = (__bf16)(xv * m2 + xx * (1.f - m2));
  }
}

// ---------------------------------------------------------------------------
// Numerically stable WKV recurrence. One thread per (batch, channel);
// coalesced along C. Fuses sigmoid(r)*wkv and emits bf16 for the Wo GEMM.
// ---------------------------------------------------------------------------
__global__ __launch_bounds__(256) void wkv_kernel(
    const float* __restrict__ k, const float* __restrict__ v,
    const float* __restrict__ r,
    const float* __restrict__ decay, const float* __restrict__ u,
    __bf16* __restrict__ srwkv) {
  const int idx = blockIdx.x * blockDim.x + threadIdx.x;   // 0 .. B*C-1
  if (idx >= Bsz * Cc) return;
  const int c  = idx % Cc;
  const int bb = idx / Cc;
  const float uc   = u[c];
  const float wdec = -__expf(decay[c]);
  float a = 0.f, bsum = 0.f, p = -1e38f;
  const long long base = (long long)bb * Nn * Cc + c;
  for (int n = 0; n < Nn; ++n) {
    const long long off = base + (long long)n * Cc;
    const float kt = k[off];
    const float vt = v[off];
    // output with current state
    const float ww = uc + kt;
    const float q  = fmaxf(p, ww);
    float e1 = __expf(p - q);
    float e2 = __expf(ww - q);
    const float out = (e1 * a + e2 * vt) / (e1 * bsum + e2);
    const float rr = r[off];
    const float sr = 1.f / (1.f + __expf(-rr));
    srwkv[off] = (__bf16)(sr * out);
    // state update
    const float ww2 = p + wdec;
    const float q2  = fmaxf(ww2, kt);
    e1 = __expf(ww2 - q2);
    e2 = __expf(kt - q2);
    a    = e1 * a + e2 * vt;
    bsum = e1 * bsum + e2;
    p    = q2;
  }
}

// ---------------------------------------------------------------------------
// bf16 WMMA GEMM:  Out[M,Nd] = A[M,K] @ W[Nd,K]^T   (fp32 accumulate)
// Block = 256 threads = 8 waves (2 along M x 4 along N).
// Wave tile 64x64: 4x4 v_wmma_f32_16x16x32_bf16 per K-step, fed by
// 4 A-frags + 4 B-frags (16 b128 loads) -> 1 load per WMMA, 4x reuse.
// Block tile 128 x 256. Fragment loads use contiguous K order, same for A and B:
//   A: lane l holds row m+l%16, 16 contiguous K values (lanes 16-31: K 16..31).
//   B: lane l holds W-row (out column) n+l%16, 16 contiguous K values.
// mode 0: fp32 store | 1: fp32 store + residual | 2: relu()^2 -> bf16 store
// ---------------------------------------------------------------------------
__global__ __launch_bounds__(256) void gemm_bf16_wmma(
    const __bf16* __restrict__ A, const __bf16* __restrict__ Wt,
    int M, int K, int Nd,
    void* __restrict__ Out, const float* __restrict__ Res, int mode) {
  const int lane  = threadIdx.x & 31;
  const int wave  = threadIdx.x >> 5;
  const int wm    = wave >> 2;          // 0..1
  const int wn    = wave & 3;           // 0..3
  const int lhalf = lane >> 4;          // 0|1
  const int l15   = lane & 15;
  const int m0 = blockIdx.y * 128 + wm * 64;
  const int n0 = blockIdx.x * 256 + wn * 64;

  v8f acc[4][4] = {};

  size_t aRow[4], bRow[4];
#pragma unroll
  for (int i = 0; i < 4; ++i) {
    aRow[i] = (size_t)(m0 + i * 16 + l15) * K;
    bRow[i] = (size_t)(n0 + i * 16 + l15) * K;
  }

  for (int kk = 0; kk < K; kk += 32) {
    const int ka = kk + lhalf * 16;   // A chunk base (lanes 16-31: K 16..31)
    const int kb = kk + lhalf * 16;   // B chunk base (lanes 16-31: K 16..31)

    v16bf a[4], b[4];
#pragma unroll
    for (int i = 0; i < 4; ++i) {
      v8bf lo = *(const v8bf*)(A + aRow[i] + ka);
      v8bf hi = *(const v8bf*)(A + aRow[i] + ka + 8);
#pragma unroll
      for (int t = 0; t < 8; ++t) { a[i][t] = lo[t]; a[i][8 + t] = hi[t]; }
    }
#pragma unroll
    for (int j = 0; j < 4; ++j)
      b[j] = *(const v16bf*)(Wt + bRow[j] + kb);

#pragma unroll
    for (int j = 0; j < 4; ++j) {
#pragma unroll
      for (int i = 0; i < 4; ++i) {
        acc[i][j] = __builtin_amdgcn_wmma_f32_16x16x32_bf16(
            false, a[i], false, b[j], (short)0, acc[i][j], false, false);
      }
    }
  }

  // Epilogue: C/D layout = VGPR r -> row base + lhalf*8 + r, col base + l15.
#pragma unroll
  for (int i = 0; i < 4; ++i) {
#pragma unroll
    for (int j = 0; j < 4; ++j) {
      const int orow = m0 + i * 16 + lhalf * 8;
      const int ocol = n0 + j * 16 + l15;
#pragma unroll
      for (int rr = 0; rr < 8; ++rr) {
        const size_t off = (size_t)(orow + rr) * Nd + ocol;
        const float vacc = acc[i][j][rr];
        if (mode == 0) {
          ((float*)Out)[off] = vacc;
        } else if (mode == 1) {
          ((float*)Out)[off] = vacc + Res[off];
        } else {
          const float rl = vacc > 0.f ? vacc : 0.f;
          ((__bf16*)Out)[off] = (__bf16)(rl * rl);
        }
      }
    }
  }
}

// ---------------------------------------------------------------------------
// Final combine: out = x1 + sigmoid(rr)*kv + x_orig
// (channel-mix residual + outer wrapper residual)
// ---------------------------------------------------------------------------
__global__ __launch_bounds__(256) void final_kernel(
    const float* __restrict__ x1, const float* __restrict__ rr,
    const float* __restrict__ kv, const float* __restrict__ x0,
    float* __restrict__ out) {
  long long i = (long long)blockIdx.x * blockDim.x + threadIdx.x;
  if (i >= BNCll) return;
  const float s = 1.f / (1.f + __expf(-rr[i]));
  out[i] = x1[i] + s * kv[i] + x0[i];
}

// ---------------------------------------------------------------------------
// Host-side orchestration
// ---------------------------------------------------------------------------
extern "C" void kernel_launch(void* const* d_in, const int* in_sizes, int n_in,
                              void* d_out, int out_size, void* d_ws, size_t ws_size,
                              hipStream_t stream) {
  (void)in_sizes; (void)n_in; (void)out_size; (void)ws_size;

  const float* x     = (const float*)d_in[0];
  const float* ln1_g = (const float*)d_in[1];
  const float* ln1_b = (const float*)d_in[2];
  const float* mu_k  = (const float*)d_in[3];
  const float* mu_v  = (const float*)d_in[4];
  const float* mu_r  = (const float*)d_in[5];
  const float* Wk    = (const float*)d_in[6];
  const float* Wv    = (const float*)d_in[7];
  const float* Wr    = (const float*)d_in[8];
  const float* Wo    = (const float*)d_in[9];
  const float* decay = (const float*)d_in[10];
  const float* u     = (const float*)d_in[11];
  const float* ln2_g = (const float*)d_in[12];
  const float* ln2_b = (const float*)d_in[13];
  const float* cmu_k = (const float*)d_in[14];
  const float* cmu_r = (const float*)d_in[15];
  const float* cWk   = (const float*)d_in[16];
  const float* cWv   = (const float*)d_in[17];
  const float* cWr   = (const float*)d_in[18];
  float* out = (float*)d_out;

  // ---- workspace carve-up ----
  char* ws = (char*)d_ws;
  size_t off = 0;
  auto take = [&](size_t bytes) -> void* {
    void* p = ws + off;
    off += (bytes + 255) & ~(size_t)255;
    return p;
  };
  const size_t CCb  = (size_t)Cc * Cc * 2;     // 2 MB bf16
  const size_t CHb  = (size_t)Cc * HID * 2;    // 8 MB bf16
  const size_t BNCf = (size_t)BNCll * 4;       // 64 MB fp32
  const size_t BNCh = (size_t)BNCll * 2;       // 32 MB bf16
  const size_t BNHh = (size_t)BNHll * 2;       // 128 MB bf16

  __bf16* wk_bf  = (__bf16*)take(CCb);
  __bf16* wv_bf  = (__bf16*)take(CCb);
  __bf16* wr_bf  = (__bf16*)take(CCb);
  __bf16* wo_bf  = (__bf16*)take(CCb);
  __bf16* cwk_bf = (__bf16*)take(CHb);
  __bf16* cwv_bf = (__bf16*)take(CHb);
  __bf16* cwr_bf = (__bf16*)take(CCb);
  float*  xs     = (float*) take(BNCf);   // LN1 out; reused as LN2 out (xc)
  __bf16* mixA   = (__bf16*)take(BNCh);   // xk  / xk2
  __bf16* mixB   = (__bf16*)take(BNCh);   // xv  / xr2
  __bf16* mixC   = (__bf16*)take(BNCh);   // xr
  float*  kbuf   = (float*) take(BNCf);   // k   ; reused as kv
  float*  vbuf   = (float*) take(BNCf);   // v   ; reused as rr (cWr out)
  float*  rbuf   = (float*) take(BNCf);   // r
  __bf16* srwkv  = (__bf16*)take(BNCh);
  float*  x1     = (float*) take(BNCf);   // x + att
  __bf16* kk_bf  = (__bf16*)take(BNHh);   // relu(xk2@cWk^T)^2 in bf16

  const dim3 blk(256);
  const dim3 cvtCC((unsigned)((Cc * (long long)Cc + 255) / 256));
  const dim3 cvtCH((unsigned)((Cc * (long long)HID + 255) / 256));
  const dim3 rows((unsigned)(Bsz * Nn));                    // 16384 LN rows
  const dim3 elems((unsigned)((BNCll + 255) / 256));        // 65536 blocks
  const dim3 wkvg((unsigned)((Bsz * Cc + 255) / 256));      // 64 blocks
  const int  M = Bsz * Nn;                                  // 16384

  // 0) weights -> bf16
  cvt_f32_bf16<<<cvtCC, blk, 0, stream>>>(Wk,  wk_bf,  (long long)Cc * Cc);
  cvt_f32_bf16<<<cvtCC, blk, 0, stream>>>(Wv,  wv_bf,  (long long)Cc * Cc);
  cvt_f32_bf16<<<cvtCC, blk, 0, stream>>>(Wr,  wr_bf,  (long long)Cc * Cc);
  cvt_f32_bf16<<<cvtCC, blk, 0, stream>>>(Wo,  wo_bf,  (long long)Cc * Cc);
  cvt_f32_bf16<<<cvtCH, blk, 0, stream>>>(cWk, cwk_bf, (long long)Cc * HID);
  cvt_f32_bf16<<<cvtCH, blk, 0, stream>>>(cWv, cwv_bf, (long long)Cc * HID);
  cvt_f32_bf16<<<cvtCC, blk, 0, stream>>>(cWr, cwr_bf, (long long)Cc * Cc);

  // ---- spatial mix ----
  ln_kernel<<<rows, blk, 0, stream>>>(x, ln1_g, ln1_b, xs);
  shiftmix_kernel<<<elems, blk, 0, stream>>>(xs, mu_k, mu_v, mu_r,
                                             mixA, mixB, mixC);
  {
    dim3 g(Cc / 256, M / 128);  // (4, 128)
    gemm_bf16_wmma<<<g, blk, 0, stream>>>(mixA, wk_bf, M, Cc, Cc, kbuf, nullptr, 0);
    gemm_bf16_wmma<<<g, blk, 0, stream>>>(mixB, wv_bf, M, Cc, Cc, vbuf, nullptr, 0);
    gemm_bf16_wmma<<<g, blk, 0, stream>>>(mixC, wr_bf, M, Cc, Cc, rbuf, nullptr, 0);
  }
  wkv_kernel<<<wkvg, blk, 0, stream>>>(kbuf, vbuf, rbuf, decay, u, srwkv);
  {
    dim3 g(Cc / 256, M / 128);
    // x1 = x + (sigmoid(r)*wkv) @ Wo^T
    gemm_bf16_wmma<<<g, blk, 0, stream>>>(srwkv, wo_bf, M, Cc, Cc, x1, x, 1);
  }

  // ---- channel mix ----
  ln_kernel<<<rows, blk, 0, stream>>>(x1, ln2_g, ln2_b, xs);   // xs now = xc
  shiftmix_kernel<<<elems, blk, 0, stream>>>(xs, cmu_k, cmu_r, nullptr,
                                             mixA, mixB, nullptr);
  {
    dim3 gk(HID / 256, M / 128);  // (16, 128)
    // kk = relu(xk2 @ cWk^T)^2  -> bf16
    gemm_bf16_wmma<<<gk, blk, 0, stream>>>(mixA, cwk_bf, M, Cc, HID,
                                           kk_bf, nullptr, 2);
    dim3 gv(Cc / 256, M / 128);
    // kv = kk @ cWv^T  (cWv is [C, HID] -> exactly W[Nd,K] form)
    gemm_bf16_wmma<<<gv, blk, 0, stream>>>(kk_bf, cwv_bf, M, HID, Cc,
                                           kbuf, nullptr, 0);
    // rr = xr2 @ cWr^T
    gemm_bf16_wmma<<<gv, blk, 0, stream>>>(mixB, cwr_bf, M, Cc, Cc,
                                           vbuf, nullptr, 0);
  }

  // out = x1 + sigmoid(rr)*kv + x
  final_kernel<<<elems, blk, 0, stream>>>(x1, vbuf, kbuf, x, out);
}
